// GATNet_2697239462709
// MI455X (gfx1250) — matrix operand order repaired and checked by reading, not compiled, not run on hardware
//
#include <hip/hip_runtime.h>
#include <hip/hip_bf16.h>
#include <math.h>

// ---------------- problem constants ----------------
#define IN_CH   128
#define HEADS1  8
#define HID1    32
#define HC1     (HEADS1*HID1)   // 256
#define OUT_CH  16
#define NEG_SLOPE 0.2f

typedef __attribute__((ext_vector_type(16))) _Float16 v16h;
typedef __attribute__((ext_vector_type(8)))  float    v8f;

// ---------------- helpers ----------------
__device__ __forceinline__ unsigned fenc(float f) {
    unsigned b = __float_as_uint(f);
    return (b & 0x80000000u) ? ~b : (b | 0x80000000u);
}
__device__ __forceinline__ float fdec(unsigned u) {
    unsigned b = (u & 0x80000000u) ? (u & 0x7FFFFFFFu) : ~u;
    return __uint_as_float(b);
}
#define ENC_NEG_INF 0x007FFFFFu   // fenc(-inf)

// ---------------- elementwise kernels ----------------
__global__ void cast_f32_f16(const float* __restrict__ in, _Float16* __restrict__ out, int n) {
    int i = blockIdx.x * blockDim.x + threadIdx.x;
    if (i < n) out[i] = (_Float16)in[i];
}

__global__ void fill_u32(unsigned* __restrict__ p, unsigned v, int n) {
    int i = blockIdx.x * blockDim.x + threadIdx.x;
    if (i < n) p[i] = v;
}

// out1 + b -> elu -> f16 (feeds second WMMA GEMM)
__global__ void bias_elu_cast(const float* __restrict__ in, const float* __restrict__ bias,
                              _Float16* __restrict__ outh, int n, int F) {
    int i = blockIdx.x * blockDim.x + threadIdx.x;
    if (i >= n * F) return;
    float v = in[i] + bias[i % F];
    v = (v > 0.f) ? v : (expf(v) - 1.f);
    outh[i] = (_Float16)v;
}

// final: add bias, log_softmax over 16 classes
__global__ void bias_logsoftmax16(const float* __restrict__ in, const float* __restrict__ bias,
                                  float* __restrict__ out, int n) {
    int i = blockIdx.x * blockDim.x + threadIdx.x;
    if (i >= n) return;
    float v[16];
    float mx = -INFINITY;
#pragma unroll
    for (int j = 0; j < 16; ++j) { v[j] = in[i * 16 + j] + bias[j]; mx = fmaxf(mx, v[j]); }
    float s = 0.f;
#pragma unroll
    for (int j = 0; j < 16; ++j) s += expf(v[j] - mx);
    float lse = mx + logf(s);
#pragma unroll
    for (int j = 0; j < 16; ++j) out[i * 16 + j] = v[j] - lse;
}

// ---------------- B-operand pre-pack into WMMA fragment layout ----------------
// For n-tile nt and k-step kt, lane l holds 16 contiguous halves:
//   Bp[(((nt*(K/32))+kt)*32 + l)*16 + i] = (f16) W[(kt*32 + (l>>4)*8 + (i&7) + (i>>3)*16)*Nc + nt*16 + (l&15)]
// (32x16 f16 B layout matching A: lanes 0-15 K=0..7,16..23; lanes 16-31 K=8..15,24..31; col=lane&15)
__global__ void pack_B_frag(const float* __restrict__ W, _Float16* __restrict__ Bp,
                            int K, int Nc) {
    int i = blockIdx.x * blockDim.x + threadIdx.x;
    int tot = (Nc >> 4) * (K >> 5) * 32 * 16;
    if (i >= tot) return;
    int elem = i & 15;
    int lane = (i >> 4) & 31;
    int t    = i >> 9;            // nt*(K/32) + kt
    int kt   = t % (K >> 5);
    int nt   = t / (K >> 5);
    int k    = kt * 32 + (lane >> 4) * 8 + (elem & 7) + (elem >> 3) * 16;
    int col  = nt * 16 + (lane & 15);
    Bp[i] = (_Float16)W[(size_t)k * Nc + col];
}

// ---------------- WMMA GEMM: C[M,Nc] = A[M,K](f16) @ Bp(packed f16), f32 accum -------
// M % 16 == 0, Nc % 16 == 0, K % 32 == 0 (compile-time). One wave per 16x16 tile;
// K templated so the k-loop fully unrolls into back-to-back v_wmma ops.
// B fragments are pre-packed: one contiguous 32-byte load per lane per k-step.
template <int K>
__global__ void wmma_gemm_f16f32(const _Float16* __restrict__ A,
                                 const _Float16* __restrict__ Bp,
                                 float* __restrict__ C,
                                 int Nc, int tiles) {
    int wave = blockIdx.x * blockDim.y + threadIdx.y;   // wave-uniform
    if (wave >= tiles) return;                          // whole wave exits: EXEC stays all-1
    int ntN = Nc >> 4;
    int m0 = (wave / ntN) << 4;
    int nt = wave % ntN;
    int lane = threadIdx.x;
    int row  = lane & 15;
    int grp  = lane >> 4;      // 0 or 1

    const _Float16* arow  = A + (size_t)(m0 + row) * K;
    const _Float16* bbase = Bp + (((size_t)nt * (K >> 5)) * 32 + lane) * 16;

    v8f c = {};
#pragma unroll
    for (int kb = 0; kb < K; kb += 32) {
        v16h a;
        // A fragment: 16-bit A 16x32 layout (ISA 7.12.2)
        // lanes 0-15: K=0..7 (v0-3), K=16..23 (v4-7); lanes 16-31: K=8..15, K=24..31
        int ka = kb + grp * 8;
#pragma unroll
        for (int i = 0; i < 8; ++i) a[i] = arow[ka + i];
#pragma unroll
        for (int i = 0; i < 8; ++i) a[8 + i] = arow[16 + ka + i];
        // B fragment: pre-packed, contiguous 16 halves per lane
        v16h b = *(const v16h*)(bbase + (size_t)(kb >> 5) * 32 * 16);

        c = __builtin_amdgcn_wmma_f32_16x16x32_f16(false, a, false, b,
                                                   (short)0, c, false, false);
    }
    // C/D layout: VGPR r -> row = r + 8*grp, col = lane&15
    float* crow = C + (size_t)(m0 + grp * 8) * Nc + (nt << 4) + row;
#pragma unroll
    for (int r = 0; r < 8; ++r) crow[(size_t)r * Nc] = c[r];
}

// ---------------- attention kernels ----------------
// per-node logits: asrc[n,h] = <h[n,h,:], a_src[h,:]>, adst likewise
__global__ void node_logits(const float* __restrict__ h, const float* __restrict__ a_src,
                            const float* __restrict__ a_dst, float* __restrict__ asrc,
                            float* __restrict__ adst, int n, int H, int C) {
    int i = blockIdx.x * blockDim.x + threadIdx.x;
    if (i >= n * H) return;
    int node = i / H, hd = i % H;
    const float* hp = h + ((size_t)node * H + hd) * C;
    const float* as = a_src + (size_t)hd * C;
    const float* ad = a_dst + (size_t)hd * C;
    float s = 0.f, d = 0.f;
    for (int c = 0; c < C; ++c) { s += hp[c] * as[c]; d += hp[c] * ad[c]; }
    asrc[i] = s; adst[i] = d;
}

// pass 1: e = leakyrelu(asrc[src]+adst[dst]); store; segment max over dst (encoded atomicMax)
__global__ void edge_logits_max(const int* __restrict__ ei, int E, int E2,
                                const float* __restrict__ asrc, const float* __restrict__ adst,
                                float* __restrict__ ebuf, unsigned* __restrict__ emax, int H) {
    int i = blockIdx.x * blockDim.x + threadIdx.x;
    if (i >= E2 * H) return;
    int e = i / H, hd = i % H;
    int s, d;
    if (e < E) { s = ei[e]; d = ei[E + e]; } else { s = d = e - E; }
    float x = asrc[s * H + hd] + adst[d * H + hd];
    x = (x > 0.f) ? x : NEG_SLOPE * x;
    ebuf[i] = x;
    atomicMax(&emax[d * H + hd], fenc(x));
}

// pass 2: p = exp(e - emax[dst]); store; segment sum
__global__ void edge_exp_sum(const int* __restrict__ ei, int E, int E2,
                             float* __restrict__ ebuf, const unsigned* __restrict__ emax,
                             float* __restrict__ denom, int H) {
    int i = blockIdx.x * blockDim.x + threadIdx.x;
    if (i >= E2 * H) return;
    int e = i / H, hd = i % H;
    int d = (e < E) ? ei[E + e] : (e - E);
    float m = fdec(emax[d * H + hd]);
    float p = expf(ebuf[i] - m);
    ebuf[i] = p;
    atomicAdd(&denom[d * H + hd], p);
}

// pass 3: out[dst,h,c..c+3] += (p/denom[dst,h]) * h[src,h,c..c+3]  (float4 gather)
__global__ void edge_aggregate4(const int* __restrict__ ei, int E, int E2,
                                const float* __restrict__ ebuf, const float* __restrict__ denom,
                                const float* __restrict__ h, float* __restrict__ out,
                                int H, int C4) {   // C4 = C/4
    int i = blockIdx.x * blockDim.x + threadIdx.x;
    int tot = E2 * H * C4;
    if (i >= tot) return;
    int c4 = i % C4;
    int t  = i / C4;
    int hd = t % H;
    int e  = t / H;
    int s, d;
    if (e < E) { s = ei[e]; d = ei[E + e]; } else { s = d = e - E; }
    float alpha = ebuf[(size_t)e * H + hd] / (denom[(size_t)d * H + hd] + 1e-16f);
    const float4 hv = *(const float4*)(h + (((size_t)s * H + hd) * C4 + c4) * 4);
    float* op = out + (((size_t)d * H + hd) * C4 + c4) * 4;
    atomicAdd(op + 0, alpha * hv.x);
    atomicAdd(op + 1, alpha * hv.y);
    atomicAdd(op + 2, alpha * hv.z);
    atomicAdd(op + 3, alpha * hv.w);
}

// ---------------- host launch ----------------
static inline char* carve(char*& p, size_t bytes) {
    char* r = p;
    p += (bytes + 255) & ~(size_t)255;
    return r;
}
static inline unsigned blks(long long n, int bs) { return (unsigned)((n + bs - 1) / bs); }

extern "C" void kernel_launch(void* const* d_in, const int* in_sizes, int n_in,
                              void* d_out, int out_size, void* d_ws, size_t ws_size,
                              hipStream_t stream) {
    const float* x      = (const float*)d_in[0];
    const int*   ei     = (const int*)  d_in[1];
    const float* W1     = (const float*)d_in[2];
    const float* a_src1 = (const float*)d_in[3];
    const float* a_dst1 = (const float*)d_in[4];
    const float* b1     = (const float*)d_in[5];
    const float* W2     = (const float*)d_in[6];
    const float* a_src2 = (const float*)d_in[7];
    const float* a_dst2 = (const float*)d_in[8];
    const float* b2     = (const float*)d_in[9];
    float* out = (float*)d_out;

    const int N  = in_sizes[0] / IN_CH;   // 10000 (multiple of 16)
    const int E  = in_sizes[1] / 2;       // 320000
    const int E2 = E + N;                 // + self loops

    // ---- workspace carve ----
    char* p = (char*)d_ws;
    _Float16* Xh    = (_Float16*)carve(p, (size_t)N * IN_CH * 2);
    _Float16* Bp1   = (_Float16*)carve(p, (size_t)IN_CH * HC1 * 2);     // packed W1 frags
    float*    h1    = (float*)   carve(p, (size_t)N * HC1 * 4);
    float*    asrc1 = (float*)   carve(p, (size_t)N * HEADS1 * 4);
    float*    adst1 = (float*)   carve(p, (size_t)N * HEADS1 * 4);
    unsigned* emax1 = (unsigned*)carve(p, (size_t)N * HEADS1 * 4);
    float*    den1  = (float*)   carve(p, (size_t)N * HEADS1 * 4);
    float*    ebuf1 = (float*)   carve(p, (size_t)E2 * HEADS1 * 4);
    float*    out1  = (float*)   carve(p, (size_t)N * HC1 * 4);
    _Float16* h1bh  = (_Float16*)carve(p, (size_t)N * HC1 * 2);
    _Float16* Bp2   = (_Float16*)carve(p, (size_t)HC1 * OUT_CH * 2);    // packed W2 frags
    float*    h2    = (float*)   carve(p, (size_t)N * OUT_CH * 4);
    float*    asrc2 = (float*)   carve(p, (size_t)N * 4);
    float*    adst2 = (float*)   carve(p, (size_t)N * 4);
    unsigned* emax2 = (unsigned*)carve(p, (size_t)N * 4);
    float*    den2  = (float*)   carve(p, (size_t)N * 4);
    float*    ebuf2 = (float*)   carve(p, (size_t)E2 * 4);
    float*    out2  = (float*)   carve(p, (size_t)N * OUT_CH * 4);

    const int BS = 256;

    // ---- layer 1 ----
    cast_f32_f16<<<blks((long long)N * IN_CH, BS), BS, 0, stream>>>(x, Xh, N * IN_CH);
    pack_B_frag<<<blks(IN_CH * HC1, BS), BS, 0, stream>>>(W1, Bp1, IN_CH, HC1);

    {   // h1 = Xh @ W1   (M=N, K=128, Nc=256), fully unrolled 4x WMMA per tile
        int tiles = (N / 16) * (HC1 / 16);
        dim3 blk(32, 8);
        wmma_gemm_f16f32<IN_CH><<<blks(tiles, 8), blk, 0, stream>>>(Xh, Bp1, h1, HC1, tiles);
    }

    node_logits<<<blks((long long)N * HEADS1, BS), BS, 0, stream>>>(
        h1, a_src1, a_dst1, asrc1, adst1, N, HEADS1, HID1);

    fill_u32<<<blks((long long)N * HEADS1, BS), BS, 0, stream>>>(emax1, ENC_NEG_INF, N * HEADS1);
    hipMemsetAsync(den1, 0, (size_t)N * HEADS1 * 4, stream);
    hipMemsetAsync(out1, 0, (size_t)N * HC1 * 4, stream);

    edge_logits_max<<<blks((long long)E2 * HEADS1, BS), BS, 0, stream>>>(
        ei, E, E2, asrc1, adst1, ebuf1, emax1, HEADS1);
    edge_exp_sum<<<blks((long long)E2 * HEADS1, BS), BS, 0, stream>>>(
        ei, E, E2, ebuf1, emax1, den1, HEADS1);
    edge_aggregate4<<<blks((long long)E2 * HEADS1 * (HID1 / 4), BS), BS, 0, stream>>>(
        ei, E, E2, ebuf1, den1, h1, out1, HEADS1, HID1 / 4);

    // elu(out1 + b1) -> f16 features for layer 2
    bias_elu_cast<<<blks((long long)N * HC1, BS), BS, 0, stream>>>(out1, b1, h1bh, N, HC1);

    // ---- layer 2 ----
    pack_B_frag<<<blks(HC1 * OUT_CH, BS), BS, 0, stream>>>(W2, Bp2, HC1, OUT_CH);
    {   // h2 = elu(h1) @ W2  (M=N, K=256, Nc=16), fully unrolled 8x WMMA per tile
        int tiles = (N / 16) * (OUT_CH / 16);
        dim3 blk(32, 8);
        wmma_gemm_f16f32<HC1><<<blks(tiles, 8), blk, 0, stream>>>(h1bh, Bp2, h2, OUT_CH, tiles);
    }

    node_logits<<<blks((long long)N, BS), BS, 0, stream>>>(
        h2, a_src2, a_dst2, asrc2, adst2, N, 1, OUT_CH);

    fill_u32<<<blks((long long)N, BS), BS, 0, stream>>>(emax2, ENC_NEG_INF, N);
    hipMemsetAsync(den2, 0, (size_t)N * 4, stream);
    hipMemsetAsync(out2, 0, (size_t)N * OUT_CH * 4, stream);

    edge_logits_max<<<blks((long long)E2, BS), BS, 0, stream>>>(
        ei, E, E2, asrc2, adst2, ebuf2, emax2, 1);
    edge_exp_sum<<<blks((long long)E2, BS), BS, 0, stream>>>(
        ei, E, E2, ebuf2, emax2, den2, 1);
    edge_aggregate4<<<blks((long long)E2 * (OUT_CH / 4), BS), BS, 0, stream>>>(
        ei, E, E2, ebuf2, den2, h2, out2, 1, OUT_CH / 4);

    bias_logsoftmax16<<<blks((long long)N, BS), BS, 0, stream>>>(out2, b2, out, N);
}
